// GlobalMambaSequenceClassifier_1743756722398
// MI455X (gfx1250) — compile-verified, matrix operand order checked by reading, not run on hardware
//
#include <hip/hip_runtime.h>
#include <hip/hip_bf16.h>

typedef _Float16 h16;
typedef __attribute__((ext_vector_type(8)))  _Float16 v8h;
typedef __attribute__((ext_vector_type(16))) _Float16 v16h;
typedef __attribute__((ext_vector_type(8)))  float    v8f;

#define B_    4
#define L_    1024
#define DM    512
#define DI    1024
#define DS    16
#define DTR   32
#define DCONV 4
#define NTOK  (B_ * L_)   /* 4096 */

// ---------------------------------------------------------------------------
// Load one 16x32 f16 WMMA fragment (A row-major [*,K] or B=W^T with W [N,K]).
// Lanes 0-15: row=lane, K {0..7, 16..23}; lanes 16-31: row=lane-16, K {8..15,
// 24..31}; caller pre-offsets the pointer by row*K + (lane>=16 ? 8 : 0).
// ---------------------------------------------------------------------------
__device__ __forceinline__ v16h ldfrag(const h16* __restrict__ p)
{
    v8h lo = *(const v8h*)p;
    v8h hi = *(const v8h*)(p + 16);
    return __builtin_shufflevector(lo, hi, 0,1,2,3,4,5,6,7,8,9,10,11,12,13,14,15);
}

// ---------------------------------------------------------------------------
// WMMA GEMM:  C[M,N] = A[M,K] * W[N,K]^T   (A,W f16 row-major, C f32)
// One wave computes a 32x64 supertile: 2 row-tiles x 4 col-tiles, 8
// accumulators -> 8 v_wmma per K-step for 12 b128 loads (A reused 4x, B 2x).
// Requires M%32==0, N%64==0, K%32==0 (true for every call here).
// ---------------------------------------------------------------------------
__global__ __launch_bounds__(128) void wmma_gemm_f16(
    const h16* __restrict__ A, const h16* __restrict__ W,
    float* __restrict__ C, int M, int N, int K, int addC)
{
    int wave     = blockIdx.x * (blockDim.x >> 5) + (threadIdx.x >> 5);
    int tilesN   = N >> 6;
    int numTiles = (M >> 5) * tilesN;
    if (wave >= numTiles) return;              // wave-uniform: EXEC stays all-1s
    int tm = (wave / tilesN) << 5;
    int tn = (wave % tilesN) << 6;
    int lane = threadIdx.x & 31;
    int r    = lane & 15;
    int kh   = (lane >> 4) << 3;               // 0 or 8

    const h16* Ap = A + (size_t)(tm + r) * K + kh;
    const h16* Wp = W + (size_t)(tn + r) * K + kh;

    v8f acc[2][4] = {};
    for (int k = 0; k < K; k += 32) {
        v16h a0 = ldfrag(Ap + k);
        v16h a1 = ldfrag(Ap + (size_t)16 * K + k);
        v16h b0 = ldfrag(Wp + k);
        v16h b1 = ldfrag(Wp + (size_t)16 * K + k);
        v16h b2 = ldfrag(Wp + (size_t)32 * K + k);
        v16h b3 = ldfrag(Wp + (size_t)48 * K + k);
        acc[0][0] = __builtin_amdgcn_wmma_f32_16x16x32_f16(false, a0, false, b0, (short)0, acc[0][0], false, false);
        acc[0][1] = __builtin_amdgcn_wmma_f32_16x16x32_f16(false, a0, false, b1, (short)0, acc[0][1], false, false);
        acc[0][2] = __builtin_amdgcn_wmma_f32_16x16x32_f16(false, a0, false, b2, (short)0, acc[0][2], false, false);
        acc[0][3] = __builtin_amdgcn_wmma_f32_16x16x32_f16(false, a0, false, b3, (short)0, acc[0][3], false, false);
        acc[1][0] = __builtin_amdgcn_wmma_f32_16x16x32_f16(false, a1, false, b0, (short)0, acc[1][0], false, false);
        acc[1][1] = __builtin_amdgcn_wmma_f32_16x16x32_f16(false, a1, false, b1, (short)0, acc[1][1], false, false);
        acc[1][2] = __builtin_amdgcn_wmma_f32_16x16x32_f16(false, a1, false, b2, (short)0, acc[1][2], false, false);
        acc[1][3] = __builtin_amdgcn_wmma_f32_16x16x32_f16(false, a1, false, b3, (short)0, acc[1][3], false, false);
    }
    // D layout: VGPR i -> M = base + (lane<16 ? i : 8+i), N = base + (lane&15)
    int cn    = lane & 15;
    int rbase = (lane >> 4) << 3;
    #pragma unroll
    for (int i = 0; i < 2; ++i) {
        #pragma unroll
        for (int j = 0; j < 4; ++j) {
            float* Cp = C + (size_t)(tm + i * 16 + rbase) * N + (tn + j * 16 + cn);
            #pragma unroll
            for (int rr = 0; rr < 8; ++rr) {
                if (addC) Cp[(size_t)rr * N] += acc[i][j][rr];
                else      Cp[(size_t)rr * N]  = acc[i][j][rr];
            }
        }
    }
}

// ---------------------------------------------------------------------------
__global__ void f32_to_f16_k(const float* __restrict__ s, h16* __restrict__ d, int n)
{
    int i = blockIdx.x * blockDim.x + threadIdx.x;
    if (i < n) d[i] = (h16)s[i];
}

__global__ void embed_k(const int* __restrict__ x, const float* __restrict__ emb,
                        float* __restrict__ h)
{
    int i = blockIdx.x * blockDim.x + threadIdx.x;
    if (i >= NTOK * DM) return;
    int t = i / DM, d = i % DM;
    h[i] = emb[(size_t)x[t] * DM + d];
}

// one block (128 thr) per row; writes f16 and/or f32
__global__ void layernorm_k(const float* __restrict__ in, const float* __restrict__ sc,
                            const float* __restrict__ bi, h16* __restrict__ o16,
                            float* __restrict__ o32, int D)
{
    int row = blockIdx.x;
    const float* p = in + (size_t)row * D;
    __shared__ float red[128];
    float s = 0.f;
    for (int j = threadIdx.x; j < D; j += 128) s += p[j];
    red[threadIdx.x] = s; __syncthreads();
    for (int st = 64; st > 0; st >>= 1) {
        if (threadIdx.x < st) red[threadIdx.x] += red[threadIdx.x + st];
        __syncthreads();
    }
    float mean = red[0] / D; __syncthreads();
    float v = 0.f;
    for (int j = threadIdx.x; j < D; j += 128) { float t = p[j] - mean; v += t * t; }
    red[threadIdx.x] = v; __syncthreads();
    for (int st = 64; st > 0; st >>= 1) {
        if (threadIdx.x < st) red[threadIdx.x] += red[threadIdx.x + st];
        __syncthreads();
    }
    float rstd = rsqrtf(red[0] / D + 1e-5f);
    for (int j = threadIdx.x; j < D; j += 128) {
        float y = (p[j] - mean) * rstd * sc[j] + bi[j];
        if (o16) o16[(size_t)row * D + j] = (h16)y;
        if (o32) o32[(size_t)row * D + j] = y;
    }
}

// depthwise causal conv(width 4) + silu ; x = xz[:, :, 0:DI]
__global__ void conv_silu_k(const float* __restrict__ xz, const float* __restrict__ cw,
                            const float* __restrict__ cb, float* __restrict__ xo,
                            h16* __restrict__ xo16)
{
    int i = blockIdx.x * blockDim.x + threadIdx.x;
    if (i >= NTOK * DI) return;
    int d = i % DI; int t = i / DI; int l = t % L_; int b = t / L_;
    float acc = cb[d];
    #pragma unroll
    for (int j = 0; j < DCONV; ++j) {
        int ls = l - (DCONV - 1) + j;
        if (ls >= 0) acc += xz[(size_t)(b * L_ + ls) * (2 * DI) + d] * cw[d * DCONV + j];
    }
    float sv = acc / (1.f + expf(-acc));
    xo[i] = sv; xo16[i] = (h16)sv;
}

__global__ void dtr_to_f16_k(const float* __restrict__ dbc, h16* __restrict__ o)
{
    int i = blockIdx.x * blockDim.x + threadIdx.x;
    if (i >= NTOK * DTR) return;
    int t = i / DTR, r = i % DTR;
    o[i] = (h16)dbc[(size_t)t * 64 + r];
}

__global__ void softplus_k(float* __restrict__ dt, const float* __restrict__ dtb)
{
    int i = blockIdx.x * blockDim.x + threadIdx.x;
    if (i >= NTOK * DI) return;
    float x = dt[i] + dtb[i % DI];
    dt[i] = (x > 20.f) ? x : log1pf(expf(x));
}

// ---------------------------------------------------------------------------
// Selective scan. Each block = 128 channels of one batch (b is block-uniform).
// The per-timestep B/C vectors (32 floats, shared by all channels of a batch)
// are double-buffered in LDS via async global->LDS copies (ASYNCcnt-tracked),
// prefetching step l+1 while step l computes. Fuses y = ys + x*D and the
// silu(z) gate; writes f16 directly for the out_proj WMMA.
// ---------------------------------------------------------------------------
__global__ __launch_bounds__(128) void scan_k(
    const float* __restrict__ dt, const float* __restrict__ xc,
    const float* __restrict__ dbc, const float* __restrict__ Alog,
    const float* __restrict__ Dp, const float* __restrict__ xz,
    h16* __restrict__ y16)
{
    __shared__ float sBC[2][32];                 // [buf][ B(16) | C(16) ]
    int i = blockIdx.x * blockDim.x + threadIdx.x;   // exactly B_*DI threads
    int b = i / DI, d = i % DI;                  // b uniform within a block
    float A[DS], st[DS];
    #pragma unroll
    for (int s = 0; s < DS; ++s) { A[s] = -expf(Alog[d * DS + s]); st[s] = 0.f; }
    float Dv = Dp[d];

    // prefetch B/C of step 0 into buffer 0 (8 lanes x b128 = 32 floats)
    if (threadIdx.x < 8) {
        const float* g = dbc + (size_t)b * L_ * 64 + DTR + threadIdx.x * 4;
        unsigned lds = (unsigned)(uintptr_t)&sBC[0][threadIdx.x * 4];
        asm volatile("global_load_async_to_lds_b128 %0, %1, off"
                     :: "v"(lds), "v"((unsigned long long)(uintptr_t)g) : "memory");
    }

    for (int l = 0; l < L_; ++l) {
        asm volatile("s_wait_asynccnt 0x0" ::: "memory");  // prefetch landed (wave 0)
        __syncthreads();                                   // visible to all waves
        int buf = l & 1;
        if (l + 1 < L_ && threadIdx.x < 8) {               // prefetch next step
            const float* g = dbc + ((size_t)b * L_ + l + 1) * 64 + DTR + threadIdx.x * 4;
            unsigned lds = (unsigned)(uintptr_t)&sBC[buf ^ 1][threadIdx.x * 4];
            asm volatile("global_load_async_to_lds_b128 %0, %1, off"
                         :: "v"(lds), "v"((unsigned long long)(uintptr_t)g) : "memory");
        }
        size_t t = (size_t)b * L_ + l;
        float dtv = dt[t * DI + d];
        float xv  = xc[t * DI + d];
        float dtx = dtv * xv;
        float y = 0.f;
        #pragma unroll
        for (int s = 0; s < DS; ++s) {
            float dA = expf(dtv * A[s]);
            st[s] = dA * st[s] + dtx * sBC[buf][s];
            y += st[s] * sBC[buf][16 + s];
        }
        y += xv * Dv;
        float z = xz[t * (2 * DI) + DI + d];
        y *= z / (1.f + expf(-z));
        y16[t * DI + d] = (h16)y;
    }
}

__global__ void seqmean_k(const float* __restrict__ h, float* __restrict__ seq)
{
    int i = blockIdx.x * blockDim.x + threadIdx.x;
    if (i >= B_ * DM) return;
    int b = i / DM, d = i % DM;
    float s = 0.f;
    for (int l = 0; l < L_; ++l) s += h[(size_t)(b * L_ + l) * DM + d];
    seq[i] = s / (float)L_;
}

__global__ void gate_k(const float* __restrict__ seq, const float* __restrict__ mem_in,
                       const float* __restrict__ gW, const float* __restrict__ gb,
                       const float* __restrict__ uW, const float* __restrict__ ub,
                       float* __restrict__ mem_out)
{
    int i = blockIdx.x * blockDim.x + threadIdx.x;
    if (i >= B_ * DM) return;
    int b = i / DM, d = i % DM;
    const float* sb = seq + b * DM;
    const float* mb = mem_in + b * DM;
    const float* gr = gW + (size_t)d * (2 * DM);
    const float* ur = uW + (size_t)d * (2 * DM);
    float ga = gb[d], ua = ub[d];
    for (int j = 0; j < DM; ++j) { ga += sb[j] * gr[j];      ua += sb[j] * ur[j]; }
    for (int j = 0; j < DM; ++j) { ga += mb[j] * gr[DM + j]; ua += mb[j] * ur[DM + j]; }
    float g = 1.f / (1.f + expf(-ga));
    float u = tanhf(ua);
    mem_out[i] = mb[d] * (1.f - g) + u * g;
}

__global__ void addmem_k(float* __restrict__ h, const float* __restrict__ mem)
{
    int i = blockIdx.x * blockDim.x + threadIdx.x;
    if (i >= NTOK * DM) return;
    int d = i % DM; int b = (i / DM) / L_;
    h[i] += mem[b * DM + d];
}

__global__ void meminit_k(const float* __restrict__ mi, float* __restrict__ mem)
{
    int i = blockIdx.x * blockDim.x + threadIdx.x;
    if (i >= B_ * DM) return;
    mem[i] = mi[i % DM];
}

__global__ void clf_k(const float* __restrict__ pooled, const float* __restrict__ W,
                      const float* __restrict__ bias, float* __restrict__ out)
{
    int i = blockIdx.x * blockDim.x + threadIdx.x;
    if (i >= B_ * 10) return;
    int b = i / 10, c = i % 10;
    float s = bias[c];
    for (int d = 0; d < DM; ++d) s += pooled[b * DM + d] * W[c * DM + d];
    out[i] = s;
}

// ---------------------------------------------------------------------------
extern "C" void kernel_launch(void* const* d_in, const int* in_sizes, int n_in,
                              void* d_out, int out_size, void* d_ws, size_t ws_size,
                              hipStream_t stream)
{
    (void)in_sizes; (void)n_in; (void)out_size; (void)ws_size;
    const int*   x       = (const int*)  d_in[0];
    const float* emb     = (const float*)d_in[1];
    const float* nsc     = (const float*)d_in[2];
    const float* nbi     = (const float*)d_in[3];
    const float* ipW     = (const float*)d_in[4];
    const float* cW      = (const float*)d_in[5];
    const float* cb      = (const float*)d_in[6];
    const float* xpW     = (const float*)d_in[7];
    const float* dtW     = (const float*)d_in[8];
    const float* dtb     = (const float*)d_in[9];
    const float* Alog    = (const float*)d_in[10];
    const float* Dp      = (const float*)d_in[11];
    const float* opW     = (const float*)d_in[12];
    const float* gW      = (const float*)d_in[13];
    const float* gb      = (const float*)d_in[14];
    const float* uW      = (const float*)d_in[15];
    const float* ub      = (const float*)d_in[16];
    const float* memini  = (const float*)d_in[17];
    const float* fsc     = (const float*)d_in[18];
    const float* fbi     = (const float*)d_in[19];
    const float* clfW    = (const float*)d_in[20];
    const float* clfb    = (const float*)d_in[21];
    float* out = (float*)d_out;

    // workspace carve (256B aligned)
    char* w = (char*)d_ws;
    auto carve = [&](size_t bytes) { char* p = w; w += (bytes + 255) & ~(size_t)255; return (void*)p; };
    float* h      = (float*)carve((size_t)NTOK * DM * 4);
    float* xz     = (float*)carve((size_t)NTOK * 2 * DI * 4);
    float* xc     = (float*)carve((size_t)NTOK * DI * 4);   // reused for final LN (f32)
    h16*   xc16   = (h16*)  carve((size_t)NTOK * DI * 2);
    float* dbc    = (float*)carve((size_t)NTOK * 64 * 4);
    h16*   dtr16  = (h16*)  carve((size_t)NTOK * DTR * 2);
    float* dt     = (float*)carve((size_t)NTOK * DI * 4);
    h16*   y16    = (h16*)  carve((size_t)NTOK * DI * 2);
    h16*   hn16   = (h16*)  carve((size_t)NTOK * DM * 2);
    h16*   w16    = (h16*)  carve((size_t)2048 * 512 * 2);  // largest weight in f16
    float* seq    = (float*)carve((size_t)B_ * DM * 4);
    float* memA   = (float*)carve((size_t)B_ * DM * 4);
    float* memB   = (float*)carve((size_t)B_ * DM * 4);
    float* pooled = (float*)carve((size_t)B_ * DM * 4);

    auto gemm = [&](const h16* A, const h16* Wt, float* C, int M, int N, int K, int add) {
        int tiles  = (M / 32) * (N / 64);             // 32x64 supertile per wave
        int blocks = (tiles + 3) / 4;                 // 4 waves / block (128 thr, wave32)
        wmma_gemm_f16<<<blocks, 128, 0, stream>>>(A, Wt, C, M, N, K, add);
    };
    auto cvt = [&](const float* s, h16* d, int n) {
        f32_to_f16_k<<<(n + 255) / 256, 256, 0, stream>>>(s, d, n);
    };

    embed_k<<<(NTOK * DM + 255) / 256, 256, 0, stream>>>(x, emb, h);
    meminit_k<<<(B_ * DM + 127) / 128, 128, 0, stream>>>(memini, memA);
    float* memIn = memA; float* memOut = memB;

    for (int i = 0; i < 4; ++i) {
        // layernorm -> f16 activations
        layernorm_k<<<NTOK, 128, 0, stream>>>(h, nsc + i * DM, nbi + i * DM, hn16, nullptr, DM);
        // in_proj: xz = hn @ ipW^T   (4096 x 2048 x 512)
        cvt(ipW + (size_t)i * 2 * DI * DM, w16, 2 * DI * DM);
        gemm(hn16, w16, xz, NTOK, 2 * DI, DM, 0);
        // depthwise conv + silu
        conv_silu_k<<<(NTOK * DI + 255) / 256, 256, 0, stream>>>(
            xz, cW + (size_t)i * DI * DCONV, cb + (size_t)i * DI, xc, xc16);
        // x_proj: dbc = x @ xpW^T    (4096 x 64 x 1024)
        cvt(xpW + (size_t)i * 64 * DI, w16, 64 * DI);
        gemm(xc16, w16, dbc, NTOK, 64, DI, 0);
        // dt_proj: dt = softplus(dt_r @ dtW^T + dtb)   (4096 x 1024 x 32)
        dtr_to_f16_k<<<(NTOK * DTR + 255) / 256, 256, 0, stream>>>(dbc, dtr16);
        cvt(dtW + (size_t)i * DI * DTR, w16, DI * DTR);
        gemm(dtr16, w16, dt, NTOK, DI, DTR, 0);
        softplus_k<<<(NTOK * DI + 255) / 256, 256, 0, stream>>>(dt, dtb + (size_t)i * DI);
        // selective scan (async-LDS staged B/C, fused D skip + silu(z)) -> y16
        scan_k<<<B_ * DI / 128, 128, 0, stream>>>(
            dt, xc, dbc, Alog + (size_t)i * DI * DS, Dp + (size_t)i * DI, xz, y16);
        // out_proj with residual accumulate: h += y @ opW^T  (4096 x 512 x 1024)
        cvt(opW + (size_t)i * DM * DI, w16, DM * DI);
        gemm(y16, w16, h, NTOK, DM, DI, 1);
        // memory gating
        seqmean_k<<<(B_ * DM + 127) / 128, 128, 0, stream>>>(h, seq);
        gate_k<<<(B_ * DM + 127) / 128, 128, 0, stream>>>(
            seq, memIn, gW + (size_t)i * DM * 2 * DM, gb + (size_t)i * DM,
            uW + (size_t)i * DM * 2 * DM, ub + (size_t)i * DM, memOut);
        addmem_k<<<(NTOK * DM + 255) / 256, 256, 0, stream>>>(h, memOut);
        float* tmp = memIn; memIn = memOut; memOut = tmp;
    }

    // final layernorm (f32) -> mean pool -> classifier
    layernorm_k<<<NTOK, 128, 0, stream>>>(h, fsc, fbi, nullptr, xc, DM);
    seqmean_k<<<(B_ * DM + 127) / 128, 128, 0, stream>>>(xc, pooled);
    clf_k<<<1, 64, 0, stream>>>(pooled, clfW, clfb, out);
}
